// MSNetSAModule_73160472920436
// MI455X (gfx1250) — compile-verified, matrix operand order checked by reading, not run on hardware
//
#include <hip/hip_runtime.h>
#include <hip/hip_bf16.h>

typedef _Float16 half_t;
typedef __attribute__((ext_vector_type(16))) _Float16 v16h;
typedef __attribute__((ext_vector_type(8)))  _Float16 v8h;
typedef __attribute__((ext_vector_type(8)))  float    v8f;

#define BB   2
#define NN   20000
#define MM   4096
#define CC   6
#define SS   2
#define KK   25
#define JJ   64
#define RR   (BB*MM)    /* 8192 GEMM rows */
#define KC   224        /* K*Cin = 200 padded to multiple of 32 */
#define COUT 64
#define H0   128
#define H1   256

// ---------------------------------------------------------------------------
// Build xf = concat(xyz[:,:,:2], feature) as a dense (B*N, 8) f32 table.
// ---------------------------------------------------------------------------
__global__ __launch_bounds__(256) void build_xf(const float* __restrict__ xyz,
                                                const float* __restrict__ feat,
                                                float* __restrict__ xf8) {
    int i = blockIdx.x * blockDim.x + threadIdx.x;
    if (i >= BB * NN) return;
    float* dst = xf8 + (size_t)i * 8;
    dst[0] = xyz[(size_t)i * 3 + 0];
    dst[1] = xyz[(size_t)i * 3 + 1];
#pragma unroll
    for (int c = 0; c < CC; ++c) dst[2 + c] = feat[(size_t)i * CC + c];
}

// ---------------------------------------------------------------------------
// Weight prep: emit B matrices directly in WMMA fragment layout.
// Fragment element (tk, tn, lane, h) holds B[K][n] with
//   K = tk*32 + ((h&7) | ((lane>>4)<<3) | ((h>>3)<<4)),  n = tn*16 + (lane&15)
// stored contiguously: dst[((tk*NTILES+tn)*32 + lane)*16 + h]  (= dst[tid]).
// ---------------------------------------------------------------------------
__global__ __launch_bounds__(256) void prep_conv_frag(const float* __restrict__ conv_w,
                                                      half_t* __restrict__ dst) {
    int tid = blockIdx.x * blockDim.x + threadIdx.x;
    const int total = (KC / 32) * (COUT / 16) * 32 * 16;   // 14336
    if (tid >= total) return;
    int h    = tid & 15;
    int lane = (tid >> 4) & 31;
    int t    = tid >> 9;                                   // tk*NTILES + tn
    const int ntiles = COUT / 16;
    int tn = t % ntiles, tk = t / ntiles;
    int K  = tk * 32 + ((h & 7) | (((lane >> 4) & 1) << 3) | ((h >> 3) << 4));
    int n  = tn * 16 + (lane & 15);
    float v = 0.0f;
    if (K < KK * 8) {
        int c = K % 8, k = K / 8;                          // col index = k*8 + c
        v = conv_w[(size_t)n * (8 * KK) + c * KK + k];
    }
    dst[tid] = (half_t)v;
}

template <int KSRC>
__global__ __launch_bounds__(256) void prep_dense_frag(const float* __restrict__ src,
                                                       half_t* __restrict__ dst,
                                                       int Ncols) {
    int tid = blockIdx.x * blockDim.x + threadIdx.x;
    int total = (KSRC / 32) * (Ncols / 16) * 32 * 16;
    if (tid >= total) return;
    int h    = tid & 15;
    int lane = (tid >> 4) & 31;
    int t    = tid >> 9;
    int ntiles = Ncols / 16;
    int tn = t % ntiles, tk = t / ntiles;
    int K  = tk * 32 + ((h & 7) | (((lane >> 4) & 1) << 3) | ((h >> 3) << 4));
    int n  = tn * 16 + (lane & 15);
    dst[tid] = (half_t)src[(size_t)n * KSRC + K];          // src is (Ncols, KSRC)
}

// ---------------------------------------------------------------------------
// Copy new_xyz (first tuple output) into d_out.
// ---------------------------------------------------------------------------
__global__ __launch_bounds__(256) void copy_newxyz(const float* __restrict__ src,
                                                   float* __restrict__ dst) {
    int i = blockIdx.x * blockDim.x + threadIdx.x;
    if (i < BB * MM * 3) dst[i] = src[i];
}

// ---------------------------------------------------------------------------
// Gather + weighted sum: one wave32 per (s,b,m,k) unit. idx/weight streams
// (105 MB each, read once) use non-temporal loads so the L2-resident 1.28 MB
// xf table stays hot. Butterfly shfl_xor reduction; lane0 subtracts center2
// and writes 8 f16 values; k==0 waves zero pad columns [200,224).
// ---------------------------------------------------------------------------
__global__ __launch_bounds__(256) void gather_weighted(const float* __restrict__ xf8,
                                                       const float* __restrict__ wgt,
                                                       const int*   __restrict__ idx,
                                                       const float* __restrict__ new_xyz,
                                                       half_t* __restrict__ nf16) {
    const int units = SS * BB * MM * KK;             // 409600
    int wave = blockIdx.x * (blockDim.x >> 5) + (threadIdx.x >> 5);
    int lane = threadIdx.x & 31;
    if (wave >= units) return;

    int k = wave % KK; int t = wave / KK;
    int m = t % MM;    t /= MM;
    int b = t % BB;    int s = t / BB;

    const size_t base = (size_t)wave * JJ;
    float acc[8];
#pragma unroll
    for (int c = 0; c < 8; ++c) acc[c] = 0.0f;

#pragma unroll
    for (int jj = 0; jj < 2; ++jj) {
        int j  = lane + 32 * jj;
        int id = __builtin_nontemporal_load(idx + base + j);
        float w = __builtin_nontemporal_load(wgt + base + j);
        const float4* row = (const float4*)(xf8 + ((size_t)b * NN + id) * 8);
        float4 lo = row[0], hi = row[1];
        acc[0] += w * lo.x; acc[1] += w * lo.y;
        acc[2] += w * lo.z; acc[3] += w * lo.w;
        acc[4] += w * hi.x; acc[5] += w * hi.y;
        acc[6] += w * hi.z; acc[7] += w * hi.w;
    }

#pragma unroll
    for (int off = 16; off >= 1; off >>= 1) {
#pragma unroll
        for (int c = 0; c < 8; ++c) acc[c] += __shfl_xor(acc[c], off, 32);
    }

    const size_t r = (size_t)b * MM + m;             // GEMM row
    half_t* rowp = nf16 + ((size_t)s * RR + r) * KC;
    if (lane == 0) {
        acc[0] -= new_xyz[r * 3 + 0];
        acc[1] -= new_xyz[r * 3 + 1];
        half_t* dst = rowp + k * 8;
#pragma unroll
        for (int c = 0; c < 8; ++c) dst[c] = (half_t)acc[c];
    }
    if (k == 0 && lane >= 8) {                       // lanes 8..31 -> 24 pad cols
        rowp[KK * 8 + (lane - 8)] = (half_t)0.0f;
    }
}

// ---------------------------------------------------------------------------
// Wave32 WMMA GEMM: D(16x16) = relu(A(16xKDIM) * B + bias). 4 waves/block,
// each wave owns one n-tile. A row-major (lda halves); B pre-swizzled into
// fragment layout. Fully unrolled k-loop -> KDIM/32 v_wmma per wave.
// ---------------------------------------------------------------------------
template <int KDIM, int NTILES, bool OUTF32>
__global__ __launch_bounds__(128) void wmma_gemm_bias_relu(
        const half_t* __restrict__ A, int lda,
        const half_t* __restrict__ Bfrag,
        void* __restrict__ Cout_, int ldc, int colOff,
        const float* __restrict__ bias) {
    const int lane  = threadIdx.x & 31;
    const int wv    = threadIdx.x >> 5;
    const int tn    = blockIdx.y * 4 + wv;
    const int row16 = blockIdx.x * 16;
    const int col16 = tn * 16;
    const int mn    = lane & 15;                     // A row / D col
    const int khi   = (lane >> 4) & 1;

    v8f acc = {};
    const half_t* Abase = A + (size_t)(row16 + mn) * lda + khi * 8;
#pragma unroll
    for (int kt = 0; kt < KDIM / 32; ++kt) {
        v8h alo = *(const v8h*)(Abase + kt * 32);
        v8h ahi = *(const v8h*)(Abase + kt * 32 + 16);
        v16h a  = __builtin_shufflevector(alo, ahi,
                    0, 1, 2, 3, 4, 5, 6, 7, 8, 9, 10, 11, 12, 13, 14, 15);
        const half_t* bp = Bfrag + (((size_t)kt * NTILES + tn) * 32 + lane) * 16;
        v8h blo = *(const v8h*)bp;
        v8h bhi = *(const v8h*)(bp + 8);
        v16h b  = __builtin_shufflevector(blo, bhi,
                    0, 1, 2, 3, 4, 5, 6, 7, 8, 9, 10, 11, 12, 13, 14, 15);
        acc = __builtin_amdgcn_wmma_f32_16x16x32_f16(
            /*neg_a=*/false, a, /*neg_b=*/false, b,
            /*c_mod=*/(short)0, acc, /*reuse_a=*/false, /*reuse_b=*/false);
    }

    const float bv = bias[col16 + mn];
#pragma unroll
    for (int i = 0; i < 8; ++i) {
        int   mrow = i + 8 * khi;
        float v    = acc[i] + bv;
        v = v > 0.0f ? v : 0.0f;
        size_t off = (size_t)(row16 + mrow) * ldc + colOff + col16 + mn;
        if (OUTF32) ((float*)Cout_)[off] = v;
        else        ((half_t*)Cout_)[off] = (half_t)v;
    }
}

// ---------------------------------------------------------------------------
extern "C" void kernel_launch(void* const* d_in, const int* in_sizes, int n_in,
                              void* d_out, int out_size, void* d_ws, size_t ws_size,
                              hipStream_t stream) {
    const float* xyz     = (const float*)d_in[0];
    const float* feature = (const float*)d_in[1];
    const float* new_xyz = (const float*)d_in[2];
    const int*   idx     = (const int*)  d_in[3];
    const float* weight  = (const float*)d_in[4];
    const float* conv_w  = (const float*)d_in[5];
    const float* conv_b  = (const float*)d_in[6];
    const float* mlp_w0  = (const float*)d_in[7];
    const float* mlp_b0  = (const float*)d_in[8];
    const float* mlp_w1  = (const float*)d_in[9];
    const float* mlp_b1  = (const float*)d_in[10];
    float* out = (float*)d_out;

    // Workspace carve-up (256B aligned regions).
    char*  ws  = (char*)d_ws;
    size_t ofs = 0;
    auto take = [&](size_t bytes) {
        char* p = ws + ofs;
        ofs += (bytes + 255) & ~(size_t)255;
        return p;
    };
    float*  xf8  = (float*) take((size_t)BB * NN * 8 * sizeof(float));      // 1.28 MB
    half_t* nf16 = (half_t*)take((size_t)SS * RR * KC * sizeof(half_t));    // 7.34 MB
    half_t* Wc   = (half_t*)take((size_t)KC * COUT * sizeof(half_t));
    half_t* W0   = (half_t*)take((size_t)H0 * H0 * sizeof(half_t));
    half_t* W1   = (half_t*)take((size_t)H0 * H1 * sizeof(half_t));
    half_t* Y0   = (half_t*)take((size_t)RR * H0 * sizeof(half_t));         // 2 MB
    half_t* Z0   = (half_t*)take((size_t)RR * H0 * sizeof(half_t));         // 2 MB
    (void)ws_size; (void)in_sizes; (void)n_in; (void)out_size;

    // Stage 0: input packing + fragment-layout weight transposes + output #0.
    build_xf      <<<(BB * NN + 255) / 256, 256, 0, stream>>>(xyz, feature, xf8);
    prep_conv_frag<<<(KC * COUT + 255) / 256, 256, 0, stream>>>(conv_w, Wc);
    prep_dense_frag<H0><<<(H0 * H0 + 255) / 256, 256, 0, stream>>>(mlp_w0, W0, H0);
    prep_dense_frag<H0><<<(H0 * H1 + 255) / 256, 256, 0, stream>>>(mlp_w1, W1, H1);
    copy_newxyz   <<<(BB * MM * 3 + 255) / 256, 256, 0, stream>>>(new_xyz, out);

    // Stage 1: gather + weighted sum (bandwidth-dominant stage).
    {
        const int units = SS * BB * MM * KK;          // 409600 waves
        gather_weighted<<<units / 8, 256, 0, stream>>>(
            xf8, weight, idx, new_xyz, nf16);
    }

    // Stage 2: conv GEMM per scale -> Y0[r][s*64+o], f16, relu. 7 WMMA/tile.
    for (int s = 0; s < SS; ++s) {
        dim3 grid(RR / 16, (COUT / 16) / 4);          // (512, 1)
        wmma_gemm_bias_relu<KC, COUT / 16, false><<<grid, 128, 0, stream>>>(
            nf16 + (size_t)s * RR * KC, KC, Wc,
            (void*)Y0, H0, s * COUT, conv_b);
    }

    // Stage 3: MLP0 (128 -> 128), f16 out, relu. 4 WMMA/tile.
    {
        dim3 grid(RR / 16, (H0 / 16) / 4);            // (512, 2)
        wmma_gemm_bias_relu<H0, H0 / 16, false><<<grid, 128, 0, stream>>>(
            Y0, H0, W0, (void*)Z0, H0, 0, mlp_b0);
    }

    // Stage 4: MLP1 (128 -> 256), f32 out into d_out (B,M,256), relu.
    {
        dim3 grid(RR / 16, (H1 / 16) / 4);            // (512, 4)
        wmma_gemm_bias_relu<H0, H1 / 16, true><<<grid, 128, 0, stream>>>(
            Z0, H0, W1, (void*)(out + (size_t)BB * MM * 3), H1, 0, mlp_b1);
    }
}
